// FullyConnectedNetwork_88132728914334
// MI455X (gfx1250) — compile-verified, matrix operand order checked by reading, not run on hardware
//
#include <hip/hip_runtime.h>
#include <cstdint>
#include <cstddef>

// ---------------------------------------------------------------------------
// CDNA5 / gfx1250 grouped per-camera MLP:
//   pass 1: histogram cameras
//   pass 2: scan -> per-camera 128-row-padded tile offsets
//   pass 3: scatter sample ids into per-camera buckets (pad with -1)
//   pass 4: pre-swizzle W1/W2 (fp32 -> bf16 hi/lo) into WMMA B-fragment layout
//   pass 5: grouped GEMM, bf16x3 split on v_wmma_f32_16x16x32_bf16
// ---------------------------------------------------------------------------

typedef __attribute__((ext_vector_type(16))) __bf16 v16bf;
typedef __attribute__((ext_vector_type(8)))  float  v8f;

union Frag {
    v16bf    v;
    uint32_t u[8];
    uint4    q[2];
};

#define NCAM 15
#define NIN  256
#define NHID 128
#define NOUT 15

// ws layout (in int32 units):
//   [0..15]  counts   [16..31] cursors   [32..47] tileOff (16 entries, [15]=total)
//   [64.. )  idx buffer (maxTiles*128)
//   then:    fragW1 (15*8*8*2*256 dwords), fragW2 (15*4*2*256 dwords)

__device__ __forceinline__ unsigned short f32_to_bf16(float f) {
    uint32_t u = __builtin_bit_cast(uint32_t, f);
    uint32_t r = u + 0x7FFFu + ((u >> 16) & 1u);   // round-to-nearest-even
    return (unsigned short)(r >> 16);
}
__device__ __forceinline__ float bf16_to_f32(unsigned short h) {
    uint32_t u = ((uint32_t)h) << 16;
    return __builtin_bit_cast(float, u);
}
// split a,b into bf16 hi parts (packed) and bf16 residual lo parts (packed)
__device__ __forceinline__ void split_pack(float a, float b, uint32_t& hi, uint32_t& lo) {
    unsigned short ha = f32_to_bf16(a), hb = f32_to_bf16(b);
    float ra = bf16_to_f32(ha), rb = bf16_to_f32(hb);
    unsigned short la = f32_to_bf16(a - ra), lb = f32_to_bf16(b - rb);
    hi = (uint32_t)ha | ((uint32_t)hb << 16);
    lo = (uint32_t)la | ((uint32_t)lb << 16);
}

// Build an A fragment (16x32 bf16, hi+lo) for this lane from a row-major f32 row.
// A layout: lane (m + 16*h) holds row m; dwords 0-3 = K {8h..8h+7}, dwords 4-7 = +16.
__device__ __forceinline__ void build_a(Frag& ah, Frag& al,
                                        const float* rowp, int kb, bool valid) {
    float4 f0, f1, f2, f3;
    if (valid) {
        f0 = *(const float4*)(rowp + kb);
        f1 = *(const float4*)(rowp + kb + 4);
        f2 = *(const float4*)(rowp + kb + 16);
        f3 = *(const float4*)(rowp + kb + 20);
    } else {
        f0 = f1 = f2 = f3 = make_float4(0.f, 0.f, 0.f, 0.f);
    }
    split_pack(f0.x, f0.y, ah.u[0], al.u[0]);
    split_pack(f0.z, f0.w, ah.u[1], al.u[1]);
    split_pack(f1.x, f1.y, ah.u[2], al.u[2]);
    split_pack(f1.z, f1.w, ah.u[3], al.u[3]);
    split_pack(f2.x, f2.y, ah.u[4], al.u[4]);
    split_pack(f2.z, f2.w, ah.u[5], al.u[5]);
    split_pack(f3.x, f3.y, ah.u[6], al.u[6]);
    split_pack(f3.z, f3.w, ah.u[7], al.u[7]);
}

// ---------------------------- bucketing passes -----------------------------

__global__ void init_kernel(int* ws_i) {
    if (threadIdx.x < 64) ws_i[threadIdx.x] = 0;
}

__global__ void hist_kernel(const int* __restrict__ cam, int* ws_i, int B) {
    __shared__ int bins[16];
    int t = threadIdx.x;
    if (t < 16) bins[t] = 0;
    __syncthreads();
    int i = blockIdx.x * 256 + t;
    if (i < B) atomicAdd(&bins[cam[i]], 1);
    __syncthreads();
    if (t < NCAM && bins[t] > 0) atomicAdd(&ws_i[t], bins[t]);
}

__global__ void scan_kernel(int* ws_i) {
    if (threadIdx.x == 0) {
        int off = 0;
        ws_i[32] = 0;
        for (int c = 0; c < NCAM; ++c) {
            off += (ws_i[c] + 127) >> 7;          // tiles of 128 rows
            ws_i[32 + c + 1] = off;
        }
    }
}

__global__ void scatter_kernel(const int* __restrict__ cam, int* ws_i, int B) {
    int i = blockIdx.x * 256 + threadIdx.x;
    if (i >= B) return;
    int c = cam[i];
    int base = ws_i[32 + c] * 128;
    int p = atomicAdd(&ws_i[16 + c], 1);
    ws_i[64 + base + p] = i;
}

__global__ void pad_kernel(int* ws_i) {
    int c = blockIdx.x;
    int cnt = ws_i[c];
    int t0 = ws_i[32 + c], t1 = ws_i[33 + c];
    int padded = (t1 - t0) * 128;
    int p = cnt + (int)threadIdx.x;               // pad < 128 always
    if (p < padded) ws_i[64 + t0 * 128 + p] = -1;
}

// ------------------------- weight pre-swizzle passes -----------------------
// B-fragment layout mirrors A: lane (n + 16*h) holds column n; dword d, half e
// maps to k_local = (d>>2)*16 + 8*h + (d&3)*2 + e.  One fragment = 32 lanes x
// 8 dwords = 256 dwords; hi fragment at idx*256, lo fragment at idx*256+256.

__global__ void prep_w1_kernel(const float* __restrict__ W1, uint32_t* __restrict__ fragW1) {
    int t = blockIdx.x * 256 + threadIdx.x;
    if (t >= NCAM * 8 * 8 * 32 * 8) return;
    int dword = t & 7, lane = (t >> 3) & 31, kt = (t >> 8) & 7, nt = (t >> 11) & 7, c = t >> 14;
    int h = (lane >> 4) & 1, n = lane & 15;
    int kl0 = (dword >> 2) * 16 + 8 * h + (dword & 3) * 2;
    int k0 = kt * 32 + kl0;
    float v0 = W1[((size_t)(c * NIN + k0) * NHID) + nt * 16 + n];
    float v1 = W1[((size_t)(c * NIN + k0 + 1) * NHID) + nt * 16 + n];
    uint32_t hi, lo;
    split_pack(v0, v1, hi, lo);
    size_t base = ((size_t)((c * 8 + nt) * 8 + kt) * 2) * 256 + lane * 8 + dword;
    fragW1[base]       = hi;
    fragW1[base + 256] = lo;
}

__global__ void prep_w2_kernel(const float* __restrict__ W2, uint32_t* __restrict__ fragW2) {
    int t = blockIdx.x * 256 + threadIdx.x;
    if (t >= NCAM * 4 * 32 * 8) return;
    int dword = t & 7, lane = (t >> 3) & 31, kt = (t >> 8) & 3, c = t >> 10;
    int h = (lane >> 4) & 1, n = lane & 15;
    int kl0 = (dword >> 2) * 16 + 8 * h + (dword & 3) * 2;
    int k0 = kt * 32 + kl0;
    float v0 = (n < NOUT) ? W2[((size_t)(c * NHID + k0) * NOUT) + n] : 0.f;
    float v1 = (n < NOUT) ? W2[((size_t)(c * NHID + k0 + 1) * NOUT) + n] : 0.f;
    uint32_t hi, lo;
    split_pack(v0, v1, hi, lo);
    size_t base = ((size_t)(c * 4 + kt) * 2) * 256 + lane * 8 + dword;
    fragW2[base]       = hi;
    fragW2[base + 256] = lo;
}

// ------------------------------- grouped GEMM ------------------------------

#define HSTRIDE 132   // 128 + 4 pad -> conflict-free 16-row transpose via LDS

__global__ void __launch_bounds__(256)
mlp_kernel(const float* __restrict__ x,
           const float* __restrict__ b1,
           const float* __restrict__ b2,
           float* __restrict__ out,
           const int* __restrict__ ws_i,
           const uint32_t* __restrict__ fragW1,
           const uint32_t* __restrict__ fragW2) {
    __shared__ float lds_h[8 * 16 * HSTRIDE];

    const int* tileOff = ws_i + 32;
    int blk = blockIdx.x;
    if (blk >= tileOff[NCAM]) return;
    int c = 0;
    #pragma unroll
    for (int i = 1; i < NCAM; ++i) c += (blk >= tileOff[i]) ? 1 : 0;

    const int* idxbuf = ws_i + 64;
    int lane = threadIdx.x & 31;
    int wave = threadIdx.x >> 5;
    int hsel = (lane >> 4) & 1;
    int nidx = lane & 15;

    const int* myidx = idxbuf + blk * 128 + wave * 16;
    int ridx = myidx[nidx];                    // row this lane gathers for A
    bool valid = ridx >= 0;
    const float* rowp = x + (size_t)(valid ? ridx : 0) * NIN;

    v8f z = {0.f, 0.f, 0.f, 0.f, 0.f, 0.f, 0.f, 0.f};
    v8f acc[8];
    #pragma unroll
    for (int nt = 0; nt < 8; ++nt) acc[nt] = z;

    // ---- layer 1: [16x256] x [256x128], bf16x3 split ----
    const uint32_t* w1base = fragW1 + ((size_t)c * 8 * 8 * 2) * 256 + lane * 8;
    for (int kt = 0; kt < 8; ++kt) {
        Frag ah, al;
        build_a(ah, al, rowp, kt * 32 + 8 * hsel, valid);
        #pragma unroll
        for (int nt = 0; nt < 8; ++nt) {
            const uint32_t* bp = w1base + (size_t)((nt * 8 + kt) * 2) * 256;
            Frag bh, bl;
            bh.q[0] = *(const uint4*)bp;         bh.q[1] = *(const uint4*)(bp + 4);
            bl.q[0] = *(const uint4*)(bp + 256); bl.q[1] = *(const uint4*)(bp + 260);
            acc[nt] = __builtin_amdgcn_wmma_f32_16x16x32_bf16(false, ah.v, false, bh.v, (short)0, acc[nt], false, false);
            acc[nt] = __builtin_amdgcn_wmma_f32_16x16x32_bf16(false, ah.v, false, bl.v, (short)0, acc[nt], false, false);
            acc[nt] = __builtin_amdgcn_wmma_f32_16x16x32_bf16(false, al.v, false, bh.v, (short)0, acc[nt], false, false);
        }
    }

    // ---- bias + ReLU, transpose 16x128 tile through LDS ----
    float* hb = &lds_h[wave * 16 * HSTRIDE];
    #pragma unroll
    for (int nt = 0; nt < 8; ++nt) {
        int col = nt * 16 + nidx;
        float bias = b1[c * NHID + col];
        #pragma unroll
        for (int r = 0; r < 8; ++r) {
            float v = acc[nt][r] + bias;
            v = v > 0.f ? v : 0.f;
            hb[(r + 8 * hsel) * HSTRIDE + col] = v;
        }
    }
    __syncthreads();

    // ---- layer 2: [16x128] x [128x16(pad)], bf16x3 split ----
    v8f acc2 = z;
    const float* hrow = hb + nidx * HSTRIDE;
    const uint32_t* w2base = fragW2 + ((size_t)c * 4 * 2) * 256 + lane * 8;
    for (int kt = 0; kt < 4; ++kt) {
        Frag ah, al;
        build_a(ah, al, hrow, kt * 32 + 8 * hsel, true);
        const uint32_t* bp = w2base + (size_t)(kt * 2) * 256;
        Frag bh, bl;
        bh.q[0] = *(const uint4*)bp;         bh.q[1] = *(const uint4*)(bp + 4);
        bl.q[0] = *(const uint4*)(bp + 256); bl.q[1] = *(const uint4*)(bp + 260);
        acc2 = __builtin_amdgcn_wmma_f32_16x16x32_bf16(false, ah.v, false, bh.v, (short)0, acc2, false, false);
        acc2 = __builtin_amdgcn_wmma_f32_16x16x32_bf16(false, ah.v, false, bl.v, (short)0, acc2, false, false);
        acc2 = __builtin_amdgcn_wmma_f32_16x16x32_bf16(false, al.v, false, bh.v, (short)0, acc2, false, false);
    }

    // ---- bias + scatter store ----
    float bias2 = (nidx < NOUT) ? b2[c * NOUT + nidx] : 0.f;
    #pragma unroll
    for (int r = 0; r < 8; ++r) {
        int row = r + 8 * hsel;
        int oidx = myidx[row];
        if (oidx >= 0 && nidx < NOUT)
            out[(size_t)oidx * NOUT + nidx] = acc2[r] + bias2;
    }
}

// --------------------------------- launcher --------------------------------

extern "C" void kernel_launch(void* const* d_in, const int* in_sizes, int n_in,
                              void* d_out, int out_size, void* d_ws, size_t ws_size,
                              hipStream_t stream) {
    const float* x   = (const float*)d_in[0];
    const int*   cam = (const int*)  d_in[1];
    const float* W1  = (const float*)d_in[2];
    const float* b1  = (const float*)d_in[3];
    const float* W2  = (const float*)d_in[4];
    const float* b2  = (const float*)d_in[5];
    float* out = (float*)d_out;

    int B = in_sizes[0] / NIN;
    int maxTiles = (B + 127) / 128 + NCAM;     // upper bound on sum of per-camera tiles

    int* ws_i = (int*)d_ws;
    uint32_t* fragW1 = (uint32_t*)d_ws + 64 + (size_t)maxTiles * 128;
    uint32_t* fragW2 = fragW1 + (size_t)NCAM * 8 * 8 * 2 * 256;

    init_kernel   <<<1, 64, 0, stream>>>(ws_i);
    hist_kernel   <<<(B + 255) / 256, 256, 0, stream>>>(cam, ws_i, B);
    scan_kernel   <<<1, 1, 0, stream>>>(ws_i);
    scatter_kernel<<<(B + 255) / 256, 256, 0, stream>>>(cam, ws_i, B);
    pad_kernel    <<<NCAM, 128, 0, stream>>>(ws_i);

    prep_w1_kernel<<<(NCAM * 8 * 8 * 32 * 8 + 255) / 256, 256, 0, stream>>>(W1, fragW1);
    prep_w2_kernel<<<(NCAM * 4 * 32 * 8 + 255) / 256, 256, 0, stream>>>(W2, fragW2);

    mlp_kernel<<<maxTiles, 256, 0, stream>>>(x, b1, b2, out, ws_i, fragW1, fragW2);
}
